// WindowAttention_11055245819968
// MI455X (gfx1250) — compile-verified, matrix operand order checked by reading, not run on hardware
//
#include <hip/hip_runtime.h>
#include <hip/hip_bf16.h>

#define DIMC   256
#define HEADS  8
#define HD     32
#define NTOK   49
#define NPAD   64
#define NWIN   1024
#define NBATCH 8192

typedef __bf16 bf16;
typedef __attribute__((ext_vector_type(16))) __bf16 v16bf;
typedef __attribute__((ext_vector_type(8)))  float  v8f;

union FragU { uint4 q[2]; v16bf v; };

// Load a 16x32 bf16 fragment in CDNA5 WMMA A/B layout.
// lane 0..15 : row r=lane,   elems 0..7 = K 0..7,  elems 8..15 = K 16..23
// lane 16..31: row r=lane-16,elems 0..7 = K 8..15, elems 8..15 = K 24..31
__device__ __forceinline__ v16bf ld_frag(const bf16* base, int stride, int lane) {
    const int r  = lane & 15;
    const int kb = (lane >> 4) << 3;          // 0 or 8
    const bf16* p = base + r * stride + kb;
    FragU u;
    u.q[0] = *(const uint4*)(p);              // K = kb .. kb+7
    u.q[1] = *(const uint4*)(p + 16);         // K = kb+16 .. kb+23
    return u.v;
}

__device__ __forceinline__ v8f wmma_bf(v16bf a, v16bf b, v8f c) {
    return __builtin_amdgcn_wmma_f32_16x16x32_bf16(false, a, false, b,
                                                   (short)0, c, false, false);
}

__device__ __forceinline__ void lds_fence_barrier() {
    asm volatile("s_wait_dscnt 0" ::: "memory");
    __syncthreads();
}

// ---------------------------------------------------------------------------
// Prep: fp32 weights -> bf16 (reused 8192x, keep them small in L2) and
// pre-gather relative-position bias into [H][49][49] fp32.
// ---------------------------------------------------------------------------
__global__ void prep_kernel(const float* __restrict__ qkw_f,
                            const float* __restrict__ pjw_f,
                            const float* __restrict__ relb,
                            const int*   __restrict__ relidx,
                            bf16* __restrict__ qkw, bf16* __restrict__ pjw,
                            float* __restrict__ biasf) {
    int i = blockIdx.x * blockDim.x + threadIdx.x;
    if (i < 2 * DIMC * DIMC) qkw[i] = (bf16)qkw_f[i];
    if (i < DIMC * DIMC)     pjw[i] = (bf16)pjw_f[i];
    if (i < HEADS * NTOK * NTOK) {
        int h  = i / (NTOK * NTOK);
        int ij = i % (NTOK * NTOK);
        biasf[i] = relb[relidx[ij] * HEADS + h];
    }
}

// ---------------------------------------------------------------------------
// Fused window attention: one block per window, 8 waves (one head per wave
// in the attention phases).
// ---------------------------------------------------------------------------
__global__ __launch_bounds__(256) void win_attn_kernel(
    const float* __restrict__ x,  const float* __restrict__ v,
    const float* __restrict__ mask, const float* __restrict__ qk_b,
    const float* __restrict__ proj_b,
    const bf16* __restrict__ qkw, const bf16* __restrict__ pjw,
    const float* __restrict__ biasf, float* __restrict__ out)
{
    extern __shared__ char smem[];
    bf16* xs  = (bf16*)(smem);                // [64][256] x, bf16
    bf16* qs  = (bf16*)(smem + 32 * 1024);    // [64][256] q (scaled)
    bf16* ks  = (bf16*)(smem + 64 * 1024);    // [64][256] k
    bf16* vsT = (bf16*)(smem + 96 * 1024);    // [256][64] V transposed
    bf16* ps  = (bf16*)(smem + 32 * 1024);    // [8][64][64] probs (overlays qs,ks)
    bf16* os  = (bf16*)(smem);                // [64][256] attn@V      (overlays xs)

    const int b    = blockIdx.x;
    const int tid  = threadIdx.x;
    const int lane = tid & 31;
    const int wv   = tid >> 5;                // 0..7
    const int lcol = lane & 15;
    const int hi8  = (lane >> 4) * 8;         // C-layout row offset per lane half

    // ---------- Phase 1: stage x and V^T in LDS (zero-padded rows 49..63) ---
    const float* xb = x + (size_t)b * NTOK * DIMC;
    const float* vb = v + (size_t)b * NTOK * DIMC;
    for (int idx = tid; idx < NPAD * DIMC; idx += 256) {
        int row = idx >> 8, col = idx & 255;
        float xv = (row < NTOK) ? xb[row * DIMC + col] : 0.0f;
        float vv = (row < NTOK) ? vb[row * DIMC + col] : 0.0f;
        xs[idx] = (bf16)xv;
        vsT[col * NPAD + row] = (bf16)vv;
    }
    lds_fence_barrier();

    // ---------- Phase 2: qk = x @ qk_w^T + b  -> q (scaled), k -------------
    // wave wv owns output columns [wv*64, wv*64+64)
    const float scale = 0.17677669529663687f;  // hd^-0.5
    for (int cg = 0; cg < 2; ++cg) {
        v8f acc[4][2];
        #pragma unroll
        for (int mi = 0; mi < 4; ++mi)
            #pragma unroll
            for (int ci = 0; ci < 2; ++ci) acc[mi][ci] = {};
        #pragma unroll
        for (int kk = 0; kk < 8; ++kk) {
            v16bf a[4], bw[2];
            #pragma unroll
            for (int mi = 0; mi < 4; ++mi)
                a[mi] = ld_frag(xs + mi * 16 * DIMC + kk * 32, DIMC, lane);
            #pragma unroll
            for (int ci = 0; ci < 2; ++ci)
                bw[ci] = ld_frag(qkw + (size_t)(wv * 64 + cg * 32 + ci * 16) * DIMC + kk * 32,
                                 DIMC, lane);
            #pragma unroll
            for (int mi = 0; mi < 4; ++mi)
                #pragma unroll
                for (int ci = 0; ci < 2; ++ci)
                    acc[mi][ci] = wmma_bf(a[mi], bw[ci], acc[mi][ci]);
        }
        #pragma unroll
        for (int ci = 0; ci < 2; ++ci) {
            const int o  = wv * 64 + cg * 32 + ci * 16 + lcol;   // 0..511
            const float bo = qk_b[o];
            #pragma unroll
            for (int mi = 0; mi < 4; ++mi)
                #pragma unroll
                for (int j = 0; j < 8; ++j) {
                    int row = mi * 16 + j + hi8;
                    float val = acc[mi][ci][j] + bo;
                    if (o < DIMC) qs[row * DIMC + o] = (bf16)(val * scale);
                    else          ks[row * DIMC + (o - DIMC)] = (bf16)val;
                }
        }
    }
    lds_fence_barrier();

    // ---------- Phase 3: per-head attention: S = qk^T, +bias+mask, softmax -
    const int h    = wv;
    const int widx = b & (NWIN - 1);
    v16bf aq[4], bk[4];
    #pragma unroll
    for (int mi = 0; mi < 4; ++mi)
        aq[mi] = ld_frag(qs + mi * 16 * DIMC + h * HD, DIMC, lane);
    #pragma unroll
    for (int ni = 0; ni < 4; ++ni)
        bk[ni] = ld_frag(ks + ni * 16 * DIMC + h * HD, DIMC, lane);
    // all q/k reads complete before probs overwrite this LDS region
    lds_fence_barrier();

    const float* biash = biasf + h * NTOK * NTOK;
    const float* maskw = mask + (size_t)widx * NTOK * NTOK;
    bf16* psh = ps + h * NPAD * NPAD;

    for (int mi = 0; mi < 4; ++mi) {
        v8f s[4];
        #pragma unroll
        for (int ni = 0; ni < 4; ++ni) {
            s[ni] = {};
            s[ni] = wmma_bf(aq[mi], bk[ni], s[ni]);
        }
        // bias + mask, kill padded key columns
        #pragma unroll
        for (int ni = 0; ni < 4; ++ni)
            #pragma unroll
            for (int j = 0; j < 8; ++j) {
                int i  = mi * 16 + j + hi8;
                int jj = ni * 16 + lcol;
                float sv = s[ni][j];
                if (jj >= NTOK)     sv = -3.0e38f;
                else if (i < NTOK)  sv += biash[i * NTOK + jj] + maskw[i * NTOK + jj];
                s[ni][j] = sv;
            }
        // row softmax: each row lives in one 16-lane half across the 4 ni tiles
        #pragma unroll
        for (int j = 0; j < 8; ++j) {
            float m = s[0][j];
            #pragma unroll
            for (int ni = 1; ni < 4; ++ni) m = fmaxf(m, s[ni][j]);
            #pragma unroll
            for (int off = 1; off < 16; off <<= 1) m = fmaxf(m, __shfl_xor(m, off, 32));
            float sum = 0.0f;
            #pragma unroll
            for (int ni = 0; ni < 4; ++ni) {
                float p = __expf(s[ni][j] - m);
                s[ni][j] = p;
                sum += p;
            }
            #pragma unroll
            for (int off = 1; off < 16; off <<= 1) sum += __shfl_xor(sum, off, 32);
            float inv = 1.0f / sum;
            int row = mi * 16 + j + hi8;
            #pragma unroll
            for (int ni = 0; ni < 4; ++ni)
                psh[row * NPAD + ni * 16 + lcol] = (bf16)(s[ni][j] * inv);
        }
    }
    lds_fence_barrier();

    // ---------- Phase 4: O_h = P @ V_h  (per head) -------------------------
    #pragma unroll
    for (int nj = 0; nj < 2; ++nj) {
        v8f o4[4];
        #pragma unroll
        for (int mi = 0; mi < 4; ++mi) o4[mi] = {};
        #pragma unroll
        for (int kk = 0; kk < 2; ++kk) {
            v16bf bv = ld_frag(vsT + (h * HD + nj * 16) * NPAD + kk * 32, NPAD, lane);
            #pragma unroll
            for (int mi = 0; mi < 4; ++mi) {
                v16bf ap = ld_frag(psh + mi * 16 * NPAD + kk * 32, NPAD, lane);
                o4[mi] = wmma_bf(ap, bv, o4[mi]);
            }
        }
        #pragma unroll
        for (int mi = 0; mi < 4; ++mi)
            #pragma unroll
            for (int j = 0; j < 8; ++j) {
                int row = mi * 16 + j + hi8;
                os[row * DIMC + h * HD + nj * 16 + lcol] = (bf16)o4[mi][j];
            }
    }
    lds_fence_barrier();

    // ---------- Phase 5: out = O @ proj_w^T + proj_b -----------------------
    // wave wv owns output columns [wv*32, wv*32+32)
    {
        v8f acc[4][2];
        #pragma unroll
        for (int mi = 0; mi < 4; ++mi)
            #pragma unroll
            for (int ci = 0; ci < 2; ++ci) acc[mi][ci] = {};
        #pragma unroll
        for (int kk = 0; kk < 8; ++kk) {
            v16bf a[4], bw[2];
            #pragma unroll
            for (int mi = 0; mi < 4; ++mi)
                a[mi] = ld_frag(os + mi * 16 * DIMC + kk * 32, DIMC, lane);
            #pragma unroll
            for (int ci = 0; ci < 2; ++ci)
                bw[ci] = ld_frag(pjw + (size_t)(wv * 32 + ci * 16) * DIMC + kk * 32,
                                 DIMC, lane);
            #pragma unroll
            for (int mi = 0; mi < 4; ++mi)
                #pragma unroll
                for (int ci = 0; ci < 2; ++ci)
                    acc[mi][ci] = wmma_bf(a[mi], bw[ci], acc[mi][ci]);
        }
        float* outb = out + (size_t)b * NTOK * DIMC;
        #pragma unroll
        for (int ci = 0; ci < 2; ++ci) {
            const int col = wv * 32 + ci * 16 + lcol;
            const float pb = proj_b[col];
            #pragma unroll
            for (int mi = 0; mi < 4; ++mi)
                #pragma unroll
                for (int j = 0; j < 8; ++j) {
                    int row = mi * 16 + j + hi8;
                    if (row < NTOK)
                        outb[row * DIMC + col] = acc[mi][ci][j] + pb;
                }
        }
    }
}

extern "C" void kernel_launch(void* const* d_in, const int* in_sizes, int n_in,
                              void* d_out, int out_size, void* d_ws, size_t ws_size,
                              hipStream_t stream) {
    (void)in_sizes; (void)n_in; (void)out_size; (void)ws_size;
    const float* x      = (const float*)d_in[0];
    const float* v      = (const float*)d_in[1];
    const float* mask   = (const float*)d_in[2];
    const float* qk_w   = (const float*)d_in[3];
    const float* qk_b   = (const float*)d_in[4];
    const float* relb   = (const float*)d_in[5];
    const float* proj_w = (const float*)d_in[6];
    const float* proj_b = (const float*)d_in[7];
    const int*   relidx = (const int*)d_in[8];

    char* ws = (char*)d_ws;
    bf16*  qkw   = (bf16*)(ws);                 // 512*256 bf16 = 256 KB
    bf16*  pjw   = (bf16*)(ws + 262144);        // 256*256 bf16 = 128 KB
    float* biasf = (float*)(ws + 393216);       // 8*49*49 f32 ≈ 75 KB

    prep_kernel<<<(2 * DIMC * DIMC + 255) / 256, 256, 0, stream>>>(
        qk_w, proj_w, relb, relidx, qkw, pjw, biasf);

    const size_t lds_bytes = 128 * 1024;
    (void)hipFuncSetAttribute((const void*)win_attn_kernel,
                              hipFuncAttributeMaxDynamicSharedMemorySize,
                              (int)lds_bytes);
    win_attn_kernel<<<NBATCH, 256, lds_bytes, stream>>>(
        x, v, mask, qk_b, proj_b, qkw, pjw, biasf, (float*)d_out);
}
